// AttentionEncoderModel_43233140801969
// MI455X (gfx1250) — matrix operand order repaired and checked by reading, not compiled, not run on hardware
//
#include <hip/hip_runtime.h>
#include <math.h>

// ---------------------------------------------------------------------------
// AttentionEncoderModel on MI455X (gfx1250, wave32, WMMA 16x16x32 f16->f32)
//
// B=2 S=512 ; convs 3->16->16->16 (4x4, stride2, pad1) ; FLAT=2880
// D=256 NH=8 DH=32 NB=4
//
// All dense math on V_WMMA_F32_16X16X32_F16 (f32 accumulate). Weights are
// pre-packed once per launch into f16 (transposed [N][K]) so B fragments are
// two contiguous 16B loads; A fragments are two contiguous 32B f32 loads +
// convertvector. Conv/GEMM geometry is templated (no div expansion). The
// softmax runs with wave-uniform bounds and b128 LDS traffic.
// ---------------------------------------------------------------------------

typedef __attribute__((ext_vector_type(16))) _Float16 v16h;
typedef __attribute__((ext_vector_type(8)))  _Float16 v8h;
typedef __attribute__((ext_vector_type(8)))  float    v8f;
typedef __attribute__((ext_vector_type(4)))  float    v4f;

#define WMMA16(a, b, c) \
  __builtin_amdgcn_wmma_f32_16x16x32_f16(false, (a), false, (b), (short)0, (c), false, false)

__device__ __forceinline__ v16h pack16(v8f a, v8f b) {
  return __builtin_shufflevector(__builtin_convertvector(a, v8h),
                                 __builtin_convertvector(b, v8h),
                                 0, 1, 2, 3, 4, 5, 6, 7, 8, 9, 10, 11, 12, 13, 14, 15);
}
__device__ __forceinline__ v16h pack16h(v8h a, v8h b) {
  return __builtin_shufflevector(a, b, 0, 1, 2, 3, 4, 5, 6, 7, 8, 9, 10, 11, 12, 13, 14, 15);
}

__device__ __forceinline__ float gelu_exact(float v) {
  return 0.5f * v * (1.0f + erff(v * 0.70710678118654752f));
}

// A-fragment K offset (CDNA5 16-bit A 16x32 layout), lane = hi*16 + lo:
// element j -> K = hi*8 + j (+8 if j >= 8); i.e. two contiguous 8-spans.
__device__ __forceinline__ int koffA(int j, int hi) {
  return hi * 8 + j + ((j >= 8) ? 8 : 0);
}

// ---------------------------------------------------------------------------
// Weight packing (once per launch, ~7.4 MB total):
//  conv:  wt[Cout=16][Kpad] f16, zero-padded K tail
//  gemm:  Bt[N][K]          f16, transposed for contiguous fragment reads
// ---------------------------------------------------------------------------
__global__ void conv_w_pack_k(const float* __restrict__ w, _Float16* __restrict__ wt,
                              int Kreal, int Kpad) {
  const int i = blockIdx.x * blockDim.x + threadIdx.x;
  if (i < 16 * Kpad) {
    const int co = i / Kpad, k = i % Kpad;
    wt[i] = (_Float16)((k < Kreal) ? w[co * Kreal + k] : 0.f);
  }
}

__global__ void wT_pack_k(const float* __restrict__ B, _Float16* __restrict__ Bt,
                          int K, int N) {
  const int i = blockIdx.x * blockDim.x + threadIdx.x;
  if (i < K * N) {
    const int n = i / K, k = i % K;
    Bt[i] = (_Float16)B[(size_t)k * N + n];
  }
}

// ---------------------------------------------------------------------------
// Implicit-GEMM conv: 4x4, stride 2, pad 1, Cout = 16, ReLU. NCHW in/out.
// M = Nimg*OH*OW rows, N = 16 (one tile), K = Cin*16 padded to /32.
// Geometry templated: / and % by constants; branchless clamped gathers.
// ---------------------------------------------------------------------------
template <int Cin, int IH, int IW, int OH, int OW, int Kreal, int Kpad>
__global__ __launch_bounds__(32)
void conv_gemm_k(const float* __restrict__ in, const _Float16* __restrict__ wt,
                 const float* __restrict__ bias, float* __restrict__ out) {
  const int l  = threadIdx.x;
  const int lo = l & 15;
  const int hi = l >> 4;
  const int m0 = blockIdx.x * 16;

  const int row = m0 + lo;
  const int ow  = row % OW;
  const int t0  = row / OW;
  const int oh  = t0 % OH;
  const int n   = t0 / OH;

  const float* __restrict__ inb = in + (size_t)n * Cin * IH * IW;
  const _Float16* __restrict__ wrow = wt + lo * Kpad + hi * 8;

  v8f acc = {0.f, 0.f, 0.f, 0.f, 0.f, 0.f, 0.f, 0.f};

#pragma unroll 2
  for (int k0 = 0; k0 < Kpad; k0 += 32) {
    v16h a;
#pragma unroll
    for (int j = 0; j < 16; ++j) {
      const int ka = k0 + koffA(j, hi);
      const int kw = ka & 3;
      const int kh = (ka >> 2) & 3;
      const int ci = (Kreal == Kpad) ? (ka >> 4) : min(ka >> 4, Cin - 1);
      const int ih = oh * 2 - 1 + kh;
      const int iw = ow * 2 - 1 + kw;
      const bool valid = (ka < Kreal) & ((unsigned)ih < (unsigned)IH) &
                         ((unsigned)iw < (unsigned)IW);
      const int ihc = min(max(ih, 0), IH - 1);
      const int iwc = min(max(iw, 0), IW - 1);
      const float av = inb[((size_t)ci * IH + ihc) * IW + iwc];  // unconditional
      a[j] = (_Float16)(valid ? av : 0.f);
    }
    const v8h b0 = *(const v8h*)(wrow + k0);
    const v8h b1 = *(const v8h*)(wrow + k0 + 16);
    acc = WMMA16(a, pack16h(b0, b1), acc);
  }

  const float bb = bias[lo];
#pragma unroll
  for (int r = 0; r < 8; ++r) {
    const int rr  = m0 + hi * 8 + r;
    const int oww = rr % OW;
    const int tt  = rr / OW;
    const int ohh = tt % OH;
    const int nn  = tt / OH;
    float v = acc[r] + bb;
    v = v > 0.f ? v : 0.f;
    out[(((size_t)nn * 16 + lo) * OH + ohh) * OW + oww] = v;
  }
}

// ---------------------------------------------------------------------------
// Generic WMMA GEMM, NT n-tiles per wave (A fragment reused NT times):
// C[M,N] = act(A[M,K] @ B + bias) (+pos) (+residual), B packed f16 [N][K].
// act: 0 none, 1 exact GELU. grid = (M/16, N/(16*NT)), block = 32.
// A stream gets global_prefetch for the long-K (streaming) case.
// ---------------------------------------------------------------------------
template <int NT, int K>
__global__ __launch_bounds__(32)
void gemm_wmma_k(const float* __restrict__ A, const _Float16* __restrict__ Bt,
                 const float* __restrict__ bias, const float* __restrict__ residual,
                 const float* __restrict__ pos, float* __restrict__ C,
                 int N, int act) {
  const int l  = threadIdx.x;
  const int lo = l & 15;
  const int hi = l >> 4;
  const int m0 = blockIdx.x * 16;
  const int n0 = blockIdx.y * 16 * NT;

  const v8f vz = {0.f, 0.f, 0.f, 0.f, 0.f, 0.f, 0.f, 0.f};
  v8f acc[NT];
#pragma unroll
  for (int t = 0; t < NT; ++t) acc[t] = vz;

  const float* __restrict__ arow = A + (size_t)(m0 + lo) * K;
  const _Float16* brow[NT];
#pragma unroll
  for (int t = 0; t < NT; ++t)
    brow[t] = Bt + (size_t)(n0 + t * 16 + lo) * K + hi * 8;

#pragma unroll 2
  for (int k0 = 0; k0 < K; k0 += 32) {
    if constexpr (K >= 512) {       // streaming A: prefetch ~1KB ahead
      if (k0 + 256 < K) __builtin_prefetch(arow + k0 + 256, 0, 1);
    }
    const v8f a0 = *(const v8f*)(arow + k0 + hi * 8);
    const v8f a1 = *(const v8f*)(arow + k0 + 16 + hi * 8);
    const v16h a = pack16(a0, a1);
#pragma unroll
    for (int t = 0; t < NT; ++t) {
      const v8h b0 = *(const v8h*)(brow[t] + k0);
      const v8h b1 = *(const v8h*)(brow[t] + k0 + 16);
      acc[t] = WMMA16(a, pack16h(b0, b1), acc[t]);
    }
  }

#pragma unroll
  for (int t = 0; t < NT; ++t) {
    const int col = n0 + t * 16 + lo;
    const float bv = bias ? bias[col] : 0.f;
#pragma unroll
    for (int r = 0; r < 8; ++r) {
      const int row = m0 + hi * 8 + r;
      float v = acc[t][r] + bv;
      if (pos)      v += pos[(size_t)(row & 511) * N + col];
      if (act == 1) v  = gelu_exact(v);
      if (residual) v += residual[(size_t)row * N + col];
      C[(size_t)row * N + col] = v;
    }
  }
}

// ---------------------------------------------------------------------------
// LayerNorm over last dim D = 256. One wave per row. y may alias x.
// ---------------------------------------------------------------------------
__global__ __launch_bounds__(32)
void layernorm_k(const float* __restrict__ x, const float* __restrict__ g,
                 const float* __restrict__ b, float* __restrict__ y) {
  constexpr int D = 256;
  const int row = blockIdx.x;
  const int l   = threadIdx.x;
  const float* xr = x + (size_t)row * D;

  float s = 0.f, ss = 0.f;
#pragma unroll
  for (int i = l; i < D; i += 32) { const float v = xr[i]; s += v; ss += v * v; }
#pragma unroll
  for (int o = 16; o; o >>= 1) { s += __shfl_xor(s, o, 32); ss += __shfl_xor(ss, o, 32); }
  const float mu  = s * (1.0f / D);
  const float var = ss * (1.0f / D) - mu * mu;
  const float inv = rsqrtf(var + 1e-5f);
#pragma unroll
  for (int i = l; i < D; i += 32)
    y[(size_t)row * D + i] = (xr[i] - mu) * inv * g[i] + b[i];
}

// ---------------------------------------------------------------------------
// Causal multi-head attention, WMMA for QK^T and P@V.
// c = QKV buffer [1024, 768] (q|k|v, 256 each). x += attention (in place).
// grid = (S/16, NH, B), block = 32. DH = 32 -> QK^T is one wmma per tile.
// Softmax: wave-uniform bounds, per-element causal select, b128 LDS traffic.
// ---------------------------------------------------------------------------
__global__ __launch_bounds__(32)
void attention_k(const float* __restrict__ c, float* __restrict__ x) {
  __shared__ float Pl[16 * 512];
  __shared__ float rsum[16];

  const int qt   = blockIdx.x;
  const int h    = blockIdx.y;
  const int bb   = blockIdx.z;
  const int l    = threadIdx.x;
  const int lo   = l & 15;
  const int hi   = l >> 4;
  const int base = bb * 512;
  const int qt0  = qt * 16;

  // Q fragment (A-layout): two contiguous 32B spans of the head dim
  v16h qa;
  {
    const float* qrow = c + (size_t)(base + qt0 + lo) * 768 + h * 32;
    qa = pack16(*(const v8f*)(qrow + hi * 8), *(const v8f*)(qrow + 16 + hi * 8));
  }

  const int nkpad   = ((qt0 + 16 + 31) & ~31);  // keys rounded to /32, <= 512
  const float scale = 0.17677669529663688f;     // 1/sqrt(32)

  // scores into LDS (causal mask applied)
  for (int j0 = 0; j0 <= qt0; j0 += 16) {
    const float* krow = c + (size_t)(base + j0 + lo) * 768 + 256 + h * 32 + hi * 8;
    const v16h kb = pack16(*(const v8f*)krow, *(const v8f*)(krow + 16));
    v8f sc = {0.f, 0.f, 0.f, 0.f, 0.f, 0.f, 0.f, 0.f};
    sc = WMMA16(qa, kb, sc);
#pragma unroll
    for (int r = 0; r < 8; ++r) {
      const int qm = hi * 8 + r;
      float v = sc[r] * scale;
      if (j0 + lo > qt0 + qm) v = -1e30f;
      Pl[qm * 512 + j0 + lo] = v;
    }
  }
  __syncthreads();

  // softmax: lane pair (lo, lo+16) handles row lo; uniform trip count,
  // causal bound as per-element select; float4 LDS reads/writes.
  {
    const int qg = qt0 + lo;
    float* prow = &Pl[lo * 512];
    float mx = -1e30f;
    for (int i0 = 0; i0 < nkpad; i0 += 8) {
      const int idx = i0 + hi * 4;
      const v4f f = *(const v4f*)(prow + idx);
#pragma unroll
      for (int e = 0; e < 4; ++e)
        mx = fmaxf(mx, (idx + e <= qg) ? f[e] : -1e30f);
    }
    mx = fmaxf(mx, __shfl_xor(mx, 16, 32));
    float sum = 0.f;
    for (int i0 = 0; i0 < nkpad; i0 += 8) {
      const int idx = i0 + hi * 4;
      v4f f = *(const v4f*)(prow + idx);
#pragma unroll
      for (int e = 0; e < 4; ++e) {
        const float p = (idx + e <= qg) ? __expf(f[e] - mx) : 0.f;
        f[e] = p;
        sum += p;
      }
      *(v4f*)(prow + idx) = f;
    }
    sum += __shfl_xor(sum, 16, 32);
    if (hi == 0) rsum[lo] = sum;
  }
  __syncthreads();

  // O = P @ V (K = 32 keys per wmma), two 16-col tiles of the head dim
  v8f O0 = {0.f, 0.f, 0.f, 0.f, 0.f, 0.f, 0.f, 0.f};
  v8f O1 = {0.f, 0.f, 0.f, 0.f, 0.f, 0.f, 0.f, 0.f};
  for (int j0 = 0; j0 < nkpad; j0 += 32) {
    const float* pl = &Pl[lo * 512 + j0];
    const v16h pa = pack16(*(const v8f*)(pl + hi * 8), *(const v8f*)(pl + 16 + hi * 8));
    v16h vb0, vb1;
#pragma unroll
    for (int j = 0; j < 16; ++j) {
      const float* vr = c + (size_t)(base + j0 + koffA(j, hi)) * 768 + 512 + h * 32;
      vb0[j] = (_Float16)vr[lo];
      vb1[j] = (_Float16)vr[16 + lo];
    }
    O0 = WMMA16(pa, vb0, O0);
    O1 = WMMA16(pa, vb1, O1);
  }

#pragma unroll
  for (int r = 0; r < 8; ++r) {
    const int m = hi * 8 + r;
    const float inv = 1.0f / rsum[m];
    float* orow = x + (size_t)(base + qt0 + m) * 256 + h * 32;
    orow[lo]      += O0[r] * inv;
    orow[16 + lo] += O1[r] * inv;
  }
}

// ---------------------------------------------------------------------------
// Global standardization (mean / unbiased std over all elements).
// ---------------------------------------------------------------------------
__global__ void zero2_k(float* p) { if (threadIdx.x < 2) p[threadIdx.x] = 0.f; }

__global__ void reduce_k(const float* __restrict__ x, float* __restrict__ partial, int n) {
  float s = 0.f, ss = 0.f;
  for (int i = blockIdx.x * blockDim.x + threadIdx.x; i < n; i += gridDim.x * blockDim.x) {
    const float v = x[i]; s += v; ss += v * v;
  }
#pragma unroll
  for (int o = 16; o; o >>= 1) { s += __shfl_xor(s, o, 32); ss += __shfl_xor(ss, o, 32); }
  if ((threadIdx.x & 31) == 0) { atomicAdd(&partial[0], s); atomicAdd(&partial[1], ss); }
}

__global__ void standardize_k(float* __restrict__ y, const float* __restrict__ partial, int n) {
  const float mean = partial[0] / (float)n;
  const float var  = (partial[1] - partial[0] * partial[0] / (float)n) / (float)(n - 1);
  const float inv  = rsqrtf(var);
  for (int i = blockIdx.x * blockDim.x + threadIdx.x; i < n; i += gridDim.x * blockDim.x)
    y[i] = (y[i] - mean) * inv + 1e-10f;
}

// ---------------------------------------------------------------------------
// Host orchestration
// ---------------------------------------------------------------------------
extern "C" void kernel_launch(void* const* d_in, const int* in_sizes, int n_in,
                              void* d_out, int out_size, void* d_ws, size_t ws_size,
                              hipStream_t stream) {
  (void)in_sizes; (void)n_in; (void)out_size; (void)ws_size;

  const float* state   = (const float*)d_in[0];
  const float* conv_w1 = (const float*)d_in[1];
  const float* conv_b1 = (const float*)d_in[2];
  const float* conv_w2 = (const float*)d_in[3];
  const float* conv_b2 = (const float*)d_in[4];
  const float* conv_w3 = (const float*)d_in[5];
  const float* conv_b3 = (const float*)d_in[6];
  const float* pre_w   = (const float*)d_in[7];
  const float* pre_b   = (const float*)d_in[8];
  const float* pos_w   = (const float*)d_in[9];
  const float* ln1_g   = (const float*)d_in[10];
  const float* ln1_b   = (const float*)d_in[11];
  const float* enc_w   = (const float*)d_in[12];
  const float* enc_b   = (const float*)d_in[13];
  const float* ln2_g   = (const float*)d_in[14];
  const float* ln2_b   = (const float*)d_in[15];
  const float* ffn_w1  = (const float*)d_in[16];
  const float* ffn_b1  = (const float*)d_in[17];
  const float* ffn_w2  = (const float*)d_in[18];
  const float* ffn_b2  = (const float*)d_in[19];
  const float* emb_w   = (const float*)d_in[20];
  const float* emb_b   = (const float*)d_in[21];
  float* out = (float*)d_out;

  // ---- workspace: f16 packed weights in first 8 MiB, f32 activations after
  _Float16* hb     = (_Float16*)d_ws;
  _Float16* wt1h   = hb;               // 16*64      = 1024
  _Float16* wt2h   = hb + 1024;        // 16*256     = 4096
  _Float16* wt3h   = hb + 5120;        // 16*256     = 4096
  _Float16* pre_wT = hb + 9216;        // 256*2880   = 737280
  _Float16* enc_wT = hb + 746496;      // 4*768*256  = 786432
  _Float16* f1_wT  = hb + 1532928;     // 4*1024*256 = 1048576
  _Float16* f2_wT  = hb + 2581504;     // 4*256*1024 = 1048576
  _Float16* emb_wT = hb + 3630080;     // 256*256    = 65536   (end 3,695,616)

  float* fb       = (float*)((char*)d_ws + (size_t)8 * 1024 * 1024);
  float* conv1out = fb;                          // 47,185,920
  float* conv2out = fb + 47185920;               // 11,796,480
  float* conv3out = fb;                          //  2,949,120 (reuse conv1out)
  float* x        = fb + 2949120;                // 262,144
  float* xn       = fb + 2949120 + 262144;
  float* cbuf     = fb + 2949120 + 2 * 262144;   // 786,432
  float* hbuf     = fb + 2949120 + 2 * 262144 + 786432;  // 1,048,576
  float* partial  = hbuf + 1048576;              // 2 floats

  const int BS = 1024;
  dim3 blk(32);

  // ---- pack weights to f16 (conv: [16][Kpad]; gemm: transposed [N][K]) ----
  conv_w_pack_k<<<(16 * 64 + 255) / 256, 256, 0, stream>>>(conv_w1, wt1h, 48, 64);
  conv_w_pack_k<<<(16 * 256 + 255) / 256, 256, 0, stream>>>(conv_w2, wt2h, 256, 256);
  conv_w_pack_k<<<(16 * 256 + 255) / 256, 256, 0, stream>>>(conv_w3, wt3h, 256, 256);
  wT_pack_k<<<(2880 * 256 + 255) / 256, 256, 0, stream>>>(pre_w, pre_wT, 2880, 256);
  for (int k = 0; k < 4; ++k) {
    wT_pack_k<<<(256 * 768 + 255) / 256, 256, 0, stream>>>(
        enc_w + (size_t)k * 256 * 768, enc_wT + (size_t)k * 256 * 768, 256, 768);
    wT_pack_k<<<(256 * 1024 + 255) / 256, 256, 0, stream>>>(
        ffn_w1 + (size_t)k * 256 * 1024, f1_wT + (size_t)k * 256 * 1024, 256, 1024);
    wT_pack_k<<<(1024 * 256 + 255) / 256, 256, 0, stream>>>(
        ffn_w2 + (size_t)k * 1024 * 256, f2_wT + (size_t)k * 1024 * 256, 1024, 256);
  }
  wT_pack_k<<<(256 * 256 + 255) / 256, 256, 0, stream>>>(emb_w, emb_wT, 256, 256);

  // ---- conv stack (implicit-GEMM WMMA), NCHW throughout ----
  conv_gemm_k<3, 96, 120, 48, 60, 48, 64><<<(1024 * 48 * 60) / 16, blk, 0, stream>>>(
      state, wt1h, conv_b1, conv1out);
  conv_gemm_k<16, 48, 60, 24, 30, 256, 256><<<(1024 * 24 * 30) / 16, blk, 0, stream>>>(
      conv1out, wt2h, conv_b2, conv2out);
  conv_gemm_k<16, 24, 30, 12, 15, 256, 256><<<(1024 * 12 * 15) / 16, blk, 0, stream>>>(
      conv2out, wt3h, conv_b3, conv3out);

  // ---- pre projection + positional embedding ----
  gemm_wmma_k<4, 2880><<<dim3(BS / 16, 256 / 64), blk, 0, stream>>>(
      conv3out, pre_wT, pre_b, nullptr, pos_w, x, 256, 0);

  for (int k = 0; k < 4; ++k) {
    layernorm_k<<<BS, blk, 0, stream>>>(x, ln1_g + k * 256, ln1_b + k * 256, xn);
    gemm_wmma_k<4, 256><<<dim3(BS / 16, 768 / 64), blk, 0, stream>>>(
        xn, enc_wT + (size_t)k * 256 * 768, enc_b + k * 768, nullptr, nullptr,
        cbuf, 768, 0);
    attention_k<<<dim3(32, 8, 2), blk, 0, stream>>>(cbuf, x);
    layernorm_k<<<BS, blk, 0, stream>>>(x, ln2_g + k * 256, ln2_b + k * 256, x);
    gemm_wmma_k<4, 256><<<dim3(BS / 16, 1024 / 64), blk, 0, stream>>>(
        x, f1_wT + (size_t)k * 256 * 1024, ffn_b1 + k * 1024, nullptr, nullptr,
        hbuf, 1024, 1);
    gemm_wmma_k<4, 1024><<<dim3(BS / 16, 256 / 64), blk, 0, stream>>>(
        hbuf, f2_wT + (size_t)k * 1024 * 256, ffn_b2 + k * 256, x, nullptr,
        x, 256, 0);
  }

  gemm_wmma_k<4, 256><<<dim3(BS / 16, 256 / 64), blk, 0, stream>>>(
      x, emb_wT, emb_b, nullptr, nullptr, out, 256, 0);

  zero2_k<<<1, 32, 0, stream>>>(partial);
  reduce_k<<<256, 256, 0, stream>>>(out, partial, BS * 256);
  standardize_k<<<256, 256, 0, stream>>>(out, partial, BS * 256);
}
